// CRNN_2241972929133
// MI455X (gfx1250) — matrix operand order repaired and checked by reading, not compiled, not run on hardware
//
#include <hip/hip_runtime.h>
#include <hip/hip_bf16.h>
#include <math.h>

// ---------------- CDNA5 WMMA types/helpers ----------------
typedef _Float16 h16;
typedef __attribute__((ext_vector_type(16))) _Float16 v16h;
typedef __attribute__((ext_vector_type(8)))  _Float16 v8h;
typedef __attribute__((ext_vector_type(8)))  float    v8f;

__device__ __forceinline__ v16h combine16(v8h lo, v8h hi) {
    v16h r;
#pragma unroll
    for (int i = 0; i < 8; ++i) { r[i] = lo[i]; r[i + 8] = hi[i]; }
    return r;
}
__device__ __forceinline__ v16h zero16() {
    v16h r;
#pragma unroll
    for (int i = 0; i < 16; ++i) r[i] = (_Float16)0.f;
    return r;
}
// A tile (16x32 f16, row-major source, ld in halves). Lane l: M = rowb+(l&15),
// halves 0..7 = K[kb+off .. +7], halves 8..15 = K[kb+off+16 ..], off = (l>=16)?8:0.
__device__ __forceinline__ v16h loadA16(const h16* A, int ld, int rowb, int kb, int lane) {
    int m   = rowb + (lane & 15);
    int off = (lane >= 16) ? 8 : 0;
    const h16* p = A + (size_t)m * ld + kb + off;
    v8h lo = *(const v8h*)(p);
    v8h hi = *(const v8h*)(p + 16);
    return combine16(lo, hi);
}
// B tile from packed layout pack[tile][kc][lane][16 halves]: 32B-aligned per lane.
__device__ __forceinline__ v16h loadB16(const h16* P, int tile, int KC, int kc, int lane) {
    const h16* p = P + (((size_t)tile * KC + kc) * 32 + lane) * 16;
    v8h lo = *(const v8h*)(p);
    v8h hi = *(const v8h*)(p + 8);
    return combine16(lo, hi);
}
__device__ __forceinline__ v8f wmma16(v16h a, v16h b, v8f c) {
    return __builtin_amdgcn_wmma_f32_16x16x32_f16(false, a, false, b, (short)0, c, false, false);
}
__device__ __forceinline__ float sigmoidf(float x) { return 1.f / (1.f + expf(-x)); }

// ---------------- problem constants ----------------
#define BB   32
#define LL   16
#define DRNN 256
#define HS   264     // padded LDS row stride (halves): 4-bank row shift, conflict-free b128 column reads
#define M1   25088   // 512 frames * 49 positions
#define K1   768     // 3*16*16
#define N1   576
#define K2   5184    // 9*576
#define N2   64

// ---------------- weight packers ----------------
// Generic packer: W (N x K row-major f32) -> pack[N/16][Kpad/32][32][16] f16.
// Lane l: n = nt*16+(l&15); half h: k = kc*32 + ((l>=16)?8:0) + (h<8 ? h : h+8) (zero if k>=K).
__global__ void pack_wgt(const float* __restrict__ W, h16* __restrict__ P,
                         int N, int K, int Kpad) {
    int KC = Kpad >> 5;
    long total = (long)(N >> 4) * KC * 512;
    long i = (long)blockIdx.x * blockDim.x + threadIdx.x;
    if (i >= total) return;
    int hh = (int)(i & 15); long t = i >> 4;
    int lane = (int)(t & 31); t >>= 5;
    int kc = (int)(t % KC); int nt = (int)(t / KC);
    int n = nt * 16 + (lane & 15);
    int k = kc * 32 + ((lane >= 16) ? 8 : 0) + (hh < 8 ? hh : hh + 8);
    P[i] = (k < K) ? (h16)W[(size_t)n * K + k] : (h16)0.f;
}
// cnn1_w (64,576,3,3) with K-order k = tap*576 + c (tap = ky*3+kx).
__global__ void pack_w2(const float* __restrict__ W, h16* __restrict__ P) {
    const int KC = 162;
    long total = 4L * KC * 512;
    long i = (long)blockIdx.x * blockDim.x + threadIdx.x;
    if (i >= total) return;
    int hh = (int)(i & 15); long t = i >> 4;
    int lane = (int)(t & 31); t >>= 5;
    int kc = (int)(t % KC); int nt = (int)(t / KC);
    int n = nt * 16 + (lane & 15);
    int k = kc * 32 + ((lane >= 16) ? 8 : 0) + (hh < 8 ? hh : hh + 8);
    int tap = k / 576, c = k - tap * 576;
    P[i] = (h16)W[((size_t)n * 576 + c) * 9 + tap];
}

// ---------------- im2col for patchify conv ----------------
// A1[M1][K1] f16, row r = img*49 + (y*7+x), k = ch*256 + py*16 + px.
__global__ void im2col1(const float* __restrict__ frames, h16* __restrict__ A1) {
    long i = (long)blockIdx.x * blockDim.x + threadIdx.x;
    if (i >= (long)M1 * K1) return;
    int k = (int)(i % K1); long r = i / K1;
    int pos = (int)(r % 49); int n = (int)(r / 49);
    int y = pos / 7, x = pos % 7;
    int ch = k >> 8, py = (k >> 4) & 15, px = k & 15;
    A1[i] = (h16)frames[((size_t)n * 3 + ch) * 12544 + (size_t)(y * 16 + py) * 112 + (x * 16 + px)];
}

// ------- GEMM1: C1[M1][576] = A1 * W1^T + b (f16 out, NHWC), 2 N-tiles/wave -------
__global__ void gemm1(const h16* __restrict__ A1, const h16* __restrict__ B1,
                      const float* __restrict__ bias, h16* __restrict__ C1) {
    int id = blockIdx.x * 8 + (threadIdx.x >> 5);
    if (id >= (M1 / 16) * (N1 / 32)) return;    // 1568 * 18
    int lane = threadIdx.x & 31;
    int mt = id / (N1 / 32), np = id % (N1 / 32);
    int rowb = mt * 16, nt0 = np * 2;
    v8f acc0 = {}, acc1 = {};
#pragma unroll 4
    for (int kc = 0; kc < 24; ++kc) {
        v16h a  = loadA16(A1, K1, rowb, kc * 32, lane);
        v16h b0 = loadB16(B1, nt0,     24, kc, lane);
        acc0 = wmma16(a, b0, acc0);
        v16h b1 = loadB16(B1, nt0 + 1, 24, kc, lane);
        acc1 = wmma16(a, b1, acc1);
    }
    int nl = lane & 15, mh = (lane >> 4) * 8;
    float bs0 = bias[nt0 * 16 + nl];
    float bs1 = bias[nt0 * 16 + 16 + nl];
#pragma unroll
    for (int v = 0; v < 8; ++v) {
        size_t row = (size_t)(rowb + v + mh) * N1;
        C1[row + nt0 * 16 + nl]      = (h16)(acc0[v] + bs0);
        C1[row + nt0 * 16 + 16 + nl] = (h16)(acc1[v] + bs1);
    }
}

// ------- GEMM2: implicit-im2col 3x3 conv 576->64, + bias + ReLU, f32 out -------
// One wave computes the full 16x64 output tile: gathered A reused across 4 wmma.
__global__ void gemm2(const h16* __restrict__ C1, const h16* __restrict__ B2,
                      const float* __restrict__ bias, float* __restrict__ P) {
    int mt = blockIdx.x * 8 + (threadIdx.x >> 5);
    if (mt >= M1 / 16) return;                  // 1568 waves
    int lane = threadIdx.x & 31;
    int rowb = mt * 16;
    int m = rowb + (lane & 15);
    int pos = m % 49, img = m / 49;
    int y = pos / 7, x = pos % 7;
    int off = (lane >= 16) ? 8 : 0;
    v8f acc0 = {}, acc1 = {}, acc2 = {}, acc3 = {};
    for (int kc = 0; kc < 162; ++kc) {
        int kb = kc * 32;
        int tap = kb / 576;                // constant within a 32-chunk (576 % 32 == 0)
        int cb = kb - tap * 576 + off;
        int ky = tap / 3, kx = tap % 3;
        int iy = y + ky - 1, ix = x + kx - 1;
        v16h a;
        if (iy >= 0 && iy < 7 && ix >= 0 && ix < 7) {
            const h16* p = C1 + ((size_t)img * 49 + iy * 7 + ix) * N1 + cb;
            a = combine16(*(const v8h*)p, *(const v8h*)(p + 16));
        } else {
            a = zero16();
        }
        acc0 = wmma16(a, loadB16(B2, 0, 162, kc, lane), acc0);
        acc1 = wmma16(a, loadB16(B2, 1, 162, kc, lane), acc1);
        acc2 = wmma16(a, loadB16(B2, 2, 162, kc, lane), acc2);
        acc3 = wmma16(a, loadB16(B2, 3, 162, kc, lane), acc3);
    }
    int nl = lane & 15, mh = (lane >> 4) * 8;
    float bs0 = bias[nl], bs1 = bias[16 + nl], bs2 = bias[32 + nl], bs3 = bias[48 + nl];
#pragma unroll
    for (int v = 0; v < 8; ++v) {
        size_t row = (size_t)(rowb + v + mh) * N2;
        P[row + nl]      = fmaxf(acc0[v] + bs0, 0.f);
        P[row + 16 + nl] = fmaxf(acc1[v] + bs1, 0.f);
        P[row + 32 + nl] = fmaxf(acc2[v] + bs2, 0.f);
        P[row + 48 + nl] = fmaxf(acc3[v] + bs3, 0.f);
    }
}

// ---------------- pool over 49 positions + BN affine ----------------
__global__ void pool_bn(const float* __restrict__ P, const float* __restrict__ g,
                        const float* __restrict__ b, const float* __restrict__ m,
                        const float* __restrict__ v, float* __restrict__ F) {
    int i = blockIdx.x * blockDim.x + threadIdx.x;
    if (i >= 512 * 64) return;
    int ch = i & 63, img = i >> 6;
    float s = 0.f;
    for (int p = 0; p < 49; ++p) s += P[((size_t)img * 49 + p) * 64 + ch];
    s *= (1.f / 49.f);
    float inv = g[ch] * rsqrtf(v[ch] + 1e-5f);
    F[i] = (s - m[ch]) * inv + b[ch];
}

// ---------------- state adapters, build A_an [512][96] f16 ----------------
__global__ void adapter(const float* __restrict__ x, const float* __restrict__ F,
                        const float* __restrict__ a0w, const float* __restrict__ a0b,
                        const float* __restrict__ aiw, const float* __restrict__ aib,
                        h16* __restrict__ Aan) {
    int r = blockIdx.x * blockDim.x + threadIdx.x;
    if (r >= 512) return;
    float s0[16], s1[16];
#pragma unroll
    for (int j = 0; j < 16; ++j) {
        float acc = a0b[j];
        for (int k = 0; k < 12; ++k) acc += x[(size_t)r * 12 + k] * a0w[j * 12 + k];
        s0[j] = fmaxf(acc, 0.f);
    }
#pragma unroll
    for (int j = 0; j < 16; ++j) {
        float acc = aib[j];
        for (int k = 0; k < 16; ++k) acc += s0[k] * aiw[j * 16 + k];
        s1[j] = s0[j] + fmaxf(acc, 0.f);
    }
    h16* row = Aan + (size_t)r * 96;
    for (int j = 0; j < 16; ++j) row[j] = (h16)s1[j];
    for (int j = 0; j < 64; ++j) row[16 + j] = (h16)F[(size_t)r * 64 + j];
    for (int j = 0; j < 16; ++j) row[80 + j] = (h16)0.f;
}

// ---------------- GEMM an: As[512][256] = relu(Aan * an_w^T + b) (f16) ----------------
__global__ void gemm_an(const h16* __restrict__ Aan, const h16* __restrict__ Bp,
                        const float* __restrict__ bias, h16* __restrict__ As) {
    int id = blockIdx.x * 8 + (threadIdx.x >> 5);
    if (id >= 32 * 16) return;
    int lane = threadIdx.x & 31;
    int mt = id >> 4, nt = id & 15;
    int rowb = mt * 16, colb = nt * 16;
    v8f acc = {};
#pragma unroll
    for (int kc = 0; kc < 3; ++kc) {
        v16h a = loadA16(Aan, 96, rowb, kc * 32, lane);
        v16h b = loadB16(Bp, nt, 3, kc, lane);
        acc = wmma16(a, b, acc);
    }
    int nl = lane & 15, mh = (lane >> 4) * 8;
    float bs = bias[colb + nl];
#pragma unroll
    for (int v = 0; v < 8; ++v)
        As[(size_t)(rowb + v + mh) * 256 + colb + nl] = (h16)fmaxf(acc[v] + bs, 0.f);
}

// ---------- GEMM gi: all-step input gates GI[512][768] = As * w_ih^T + b_ih (f32) ----------
__global__ void gemm_gi(const h16* __restrict__ As, const h16* __restrict__ Bp,
                        const float* __restrict__ bias, float* __restrict__ GI) {
    int id = blockIdx.x * 8 + (threadIdx.x >> 5);
    if (id >= 32 * 48) return;
    int lane = threadIdx.x & 31;
    int mt = id / 48, nt = id % 48;
    int rowb = mt * 16, colb = nt * 16;
    v8f acc = {};
#pragma unroll
    for (int kc = 0; kc < 8; ++kc) {
        v16h a = loadA16(As, 256, rowb, kc * 32, lane);
        v16h b = loadB16(Bp, nt, 8, kc, lane);
        acc = wmma16(a, b, acc);
    }
    int nl = lane & 15, mh = (lane >> 4) * 8;
    float bs = bias[colb + nl];
#pragma unroll
    for (int v = 0; v < 8; ++v)
        GI[(size_t)(rowb + v + mh) * 768 + colb + nl] = acc[v] + bs;
}

// ----- GRU encoder(16) + autoregressive decoder(10): one WG, 32 waves, 1 tile/wave -----
// LDS state arrays use padded row stride HS=264 halves so the 16-row column reads
// that feed the A operand (ds b128 per lane) are bank-conflict-free.
__global__ void __launch_bounds__(1024) gru_kernel(
    const float* __restrict__ GI, const h16* __restrict__ Pih, const h16* __restrict__ Phh,
    const h16* __restrict__ Pfi, const float* __restrict__ b_ih, const float* __restrict__ b_hh,
    const float* __restrict__ fi_b, const float* __restrict__ fn_w,
    const float* __restrict__ fn_b, float* __restrict__ out) {
    __shared__ __align__(16) h16 hS[BB * HS];   // hidden state (f16, padded rows)
    __shared__ __align__(16) h16 xS[BB * HS];   // decoder input (f16, padded rows)
    int tid = threadIdx.x;
    int w = tid >> 5, lane = tid & 31;
    int mt = w >> 4, nt = w & 15;
    int rowb = mt * 16, colb = nt * 16;
    int nl = lane & 15, mh = (lane >> 4) * 8;

    float bhr = b_hh[colb + nl], bhz = b_hh[256 + colb + nl], bhn = b_hh[512 + colb + nl];
    float bir = b_ih[colb + nl], biz = b_ih[256 + colb + nl], bin = b_ih[512 + colb + nl];
    float bfi = fi_b[colb + nl];

    for (int i = tid; i < BB * HS; i += 1024) hS[i] = (h16)0.f;
    __syncthreads();

    // ---------------- encoder: 16 steps ----------------
    for (int t = 0; t < LL; ++t) {
        v8f gr = {}, gz = {}, gn = {};
#pragma unroll
        for (int kc = 0; kc < 8; ++kc) {
            v16h a = loadA16(hS, HS, rowb, kc * 32, lane);
            gr = wmma16(a, loadB16(Phh,      nt, 8, kc, lane), gr);
            gz = wmma16(a, loadB16(Phh, 16 + nt, 8, kc, lane), gz);
            gn = wmma16(a, loadB16(Phh, 32 + nt, 8, kc, lane), gn);
        }
        float hnew[8];
#pragma unroll
        for (int vv = 0; vv < 8; ++vv) {
            int m = rowb + vv + mh;
            const float* gi = GI + ((size_t)m * LL + t) * 768;
            float r = sigmoidf(gi[colb + nl]       + gr[vv] + bhr);
            float z = sigmoidf(gi[256 + colb + nl] + gz[vv] + bhz);
            float n = tanhf  (gi[512 + colb + nl] + r * (gn[vv] + bhn));
            float hold = (float)hS[m * HS + colb + nl];
            hnew[vv] = (1.f - z) * n + z * hold;
        }
        __syncthreads();
#pragma unroll
        for (int vv = 0; vv < 8; ++vv)
            hS[(rowb + vv + mh) * HS + colb + nl] = (h16)hnew[vv];
        __syncthreads();
    }

    // x_in = final hidden
    for (int i = tid; i < BB * HS; i += 1024) xS[i] = hS[i];
    __syncthreads();

    // ---------------- decoder: 10 autoregressive steps ----------------
    for (int step = 0; step < 10; ++step) {
        v8f gr = {}, gz = {}, gn = {}, qr = {}, qz = {}, qn = {};
#pragma unroll
        for (int kc = 0; kc < 8; ++kc) {
            v16h ah = loadA16(hS, HS, rowb, kc * 32, lane);
            gr = wmma16(ah, loadB16(Phh,      nt, 8, kc, lane), gr);
            gz = wmma16(ah, loadB16(Phh, 16 + nt, 8, kc, lane), gz);
            gn = wmma16(ah, loadB16(Phh, 32 + nt, 8, kc, lane), gn);
            v16h ax = loadA16(xS, HS, rowb, kc * 32, lane);
            qr = wmma16(ax, loadB16(Pih,      nt, 8, kc, lane), qr);
            qz = wmma16(ax, loadB16(Pih, 16 + nt, 8, kc, lane), qz);
            qn = wmma16(ax, loadB16(Pih, 32 + nt, 8, kc, lane), qn);
        }
        float hnew[8];
#pragma unroll
        for (int vv = 0; vv < 8; ++vv) {
            int m = rowb + vv + mh;
            float r = sigmoidf(qr[vv] + bir + gr[vv] + bhr);
            float z = sigmoidf(qz[vv] + biz + gz[vv] + bhz);
            float n = tanhf  (qn[vv] + bin + r * (gn[vv] + bhn));
            float hold = (float)hS[m * HS + colb + nl];
            hnew[vv] = (1.f - z) * n + z * hold;
        }
        __syncthreads();
#pragma unroll
        for (int vv = 0; vv < 8; ++vv)
            hS[(rowb + vv + mh) * HS + colb + nl] = (h16)hnew[vv];
        __syncthreads();

        // xr = hn + relu(hn @ fi_w^T + fi_b)
        v8f ff = {};
#pragma unroll
        for (int kc = 0; kc < 8; ++kc) {
            v16h a = loadA16(hS, HS, rowb, kc * 32, lane);
            ff = wmma16(a, loadB16(Pfi, nt, 8, kc, lane), ff);
        }
        float xr[8];
#pragma unroll
        for (int vv = 0; vv < 8; ++vv)
            xr[vv] = hnew[vv] + fmaxf(ff[vv] + bfi, 0.f);
        __syncthreads();
#pragma unroll
        for (int vv = 0; vv < 8; ++vv)
            xS[(rowb + vv + mh) * HS + colb + nl] = (h16)xr[vv];
        __syncthreads();

        // pred = tanh(xr @ fn_w^T + fn_b) : tiny N=2 head in VALU
        if (tid < 64) {
            int m = tid >> 1, o = tid & 1;
            float acc = fn_b[o];
            for (int k = 0; k < DRNN; ++k)
                acc += (float)xS[m * HS + k] * fn_w[o * DRNN + k];
            out[((size_t)step * BB + m) * 2 + o] = tanhf(acc);
        }
        __syncthreads();
    }
}

// ---------------- host-side orchestration ----------------
static inline size_t alignup(size_t x) { return (x + 255) & ~(size_t)255; }

extern "C" void kernel_launch(void* const* d_in, const int* in_sizes, int n_in,
                              void* d_out, int out_size, void* d_ws, size_t ws_size,
                              hipStream_t stream) {
    (void)in_sizes; (void)n_in; (void)out_size; (void)ws_size;
    const float* x      = (const float*)d_in[0];
    const float* frames = (const float*)d_in[1];
    const float* cnn_w  = (const float*)d_in[2];
    const float* cnn_b  = (const float*)d_in[3];
    const float* cnn1_w = (const float*)d_in[4];
    const float* cnn1_b = (const float*)d_in[5];
    const float* bn_g   = (const float*)d_in[6];
    const float* bn_b   = (const float*)d_in[7];
    const float* bn_m   = (const float*)d_in[8];
    const float* bn_v   = (const float*)d_in[9];
    const float* a0_w   = (const float*)d_in[10];
    const float* a0_b   = (const float*)d_in[11];
    const float* ai_w   = (const float*)d_in[12];
    const float* ai_b   = (const float*)d_in[13];
    const float* an_w   = (const float*)d_in[14];
    const float* an_b   = (const float*)d_in[15];
    const float* w_ih   = (const float*)d_in[16];
    const float* w_hh   = (const float*)d_in[17];
    const float* b_ih   = (const float*)d_in[18];
    const float* b_hh   = (const float*)d_in[19];
    const float* fi_w   = (const float*)d_in[20];
    const float* fi_b   = (const float*)d_in[21];
    const float* fn_w   = (const float*)d_in[22];
    const float* fn_b   = (const float*)d_in[23];
    float* out = (float*)d_out;

    // workspace carve-up
    char* p = (char*)d_ws; size_t off = 0;
    auto carve = [&](size_t bytes) { char* r = p + off; off = alignup(off + bytes); return (void*)r; };
    h16*   A1  = (h16*)  carve((size_t)M1 * K1 * 2);        // 38.5 MB
    h16*   B1  = (h16*)  carve((size_t)N1 * K1 * 2);
    h16*   C1  = (h16*)  carve((size_t)M1 * N1 * 2);        // 28.9 MB
    h16*   B2  = (h16*)  carve((size_t)N2 * K2 * 2);
    float* P   = (float*)carve((size_t)M1 * N2 * 4);        // 6.4 MB
    float* F   = (float*)carve((size_t)512 * 64 * 4);
    h16*   Aan = (h16*)  carve((size_t)512 * 96 * 2);
    h16*   Ban = (h16*)  carve((size_t)256 * 96 * 2);
    h16*   As  = (h16*)  carve((size_t)512 * 256 * 2);
    h16*   Pih = (h16*)  carve((size_t)768 * 256 * 2);
    h16*   Phh = (h16*)  carve((size_t)768 * 256 * 2);
    h16*   Pfi = (h16*)  carve((size_t)256 * 256 * 2);
    float* GI  = (float*)carve((size_t)512 * 768 * 4);

    // weight packing
    pack_wgt<<<(N1 / 16) * (K1 / 32) * 2, 256, 0, stream>>>(cnn_w, B1, N1, K1, K1);
    pack_w2 <<<(4 * 162 * 512 + 255) / 256, 256, 0, stream>>>(cnn1_w, B2);
    pack_wgt<<<(48 * 8 * 512) / 256, 256, 0, stream>>>(w_ih, Pih, 768, 256, 256);
    pack_wgt<<<(48 * 8 * 512) / 256, 256, 0, stream>>>(w_hh, Phh, 768, 256, 256);
    pack_wgt<<<(16 * 8 * 512) / 256, 256, 0, stream>>>(fi_w, Pfi, 256, 256, 256);
    pack_wgt<<<(16 * 3 * 512) / 256, 256, 0, stream>>>(an_w, Ban, 256, 80, 96);

    // CNN pipeline
    im2col1<<<((long)M1 * K1 + 255) / 256, 256, 0, stream>>>(frames, A1);
    gemm1  <<<((M1 / 16) * (N1 / 32) + 7) / 8, 256, 0, stream>>>(A1, B1, cnn_b, C1);   // 3528 blocks
    gemm2  <<<((M1 / 16) + 7) / 8, 256, 0, stream>>>(C1, B2, cnn1_b, P);               // 196 blocks
    pool_bn<<<(512 * 64) / 256, 256, 0, stream>>>(P, bn_g, bn_b, bn_m, bn_v, F);

    // adapters + an GEMM + hoisted input-gate GEMM
    adapter<<<2, 256, 0, stream>>>(x, F, a0_w, a0_b, ai_w, ai_b, Aan);
    gemm_an<<<(32 * 16) / 8, 256, 0, stream>>>(Aan, Ban, an_b, As);
    gemm_gi<<<(32 * 48) / 8, 256, 0, stream>>>(As, Pih, b_ih, GI);

    // recurrent encoder + decoder, single WG of 32 waves (WMMA per-tile)
    gru_kernel<<<1, 1024, 0, stream>>>(GI, Pih, Phh, Pfi, b_ih, b_hh, fi_b, fn_w, fn_b, out);
}
